// WordSAGE_56530359550767
// MI455X (gfx1250) — matrix operand order repaired and checked by reading, not compiled, not run on hardware
//
#include <hip/hip_runtime.h>
#include <cstdint>
#include <cstddef>

#define N_NODES 100000
#define N_EDGES 1600000
#define IN_C 64
#define HID_C 128
#define OUT_C 64
#define N_CLS 20

typedef __attribute__((ext_vector_type(16))) __bf16 v16bf;
typedef __attribute__((ext_vector_type(8)))  float  v8f;

union FragAB {
    v16bf          v;
    unsigned short h[16];
    unsigned int   u[8];
};

__device__ __forceinline__ unsigned short f2bf(float f) {
    unsigned int u = __float_as_uint(f);
    u += 0x7FFF + ((u >> 16) & 1);            // round-to-nearest-even
    return (unsigned short)(u >> 16);
}
__device__ __forceinline__ float bf2f(unsigned short h) {
    return __uint_as_float(((unsigned int)h) << 16);
}

// ---------------- degree ----------------
__global__ void deg_kernel(const long long* __restrict__ ei, float* __restrict__ deg) {
    int e = blockIdx.x * blockDim.x + threadIdx.x;
    if (e >= N_EDGES) return;
    int dst = (int)ei[N_EDGES + e];
    atomicAdd(&deg[dst], 1.0f);
}
__global__ void deg_inv_kernel(float* __restrict__ deg) {
    int i = blockIdx.x * blockDim.x + threadIdx.x;
    if (i >= N_NODES) return;
    deg[i] = 1.0f / fmaxf(deg[i], 1.0f);
}

// ---------------- edge scatter-add (layer 1: gather f32 x, 64ch) ----------------
__global__ void scatter1_kernel(const long long* __restrict__ ei,
                                const float* __restrict__ x,
                                float* __restrict__ msg) {
    long long tid = (long long)blockIdx.x * blockDim.x + threadIdx.x;
    if (tid >= (long long)N_EDGES * 16) return;          // 16 quads of 4ch
    int e = (int)(tid >> 4), q = (int)(tid & 15);
    int src = (int)ei[e];
    int dst = (int)ei[N_EDGES + e];
    const float4 v = *(const float4*)&x[(size_t)src * IN_C + q * 4];
    float* p = &msg[(size_t)dst * IN_C + q * 4];
    atomicAdd(p + 0, v.x); atomicAdd(p + 1, v.y);
    atomicAdd(p + 2, v.z); atomicAdd(p + 3, v.w);
}

// -------- edge scatter-add (layer 2: gather bf16 h1 from A2 cols 0..127) --------
__global__ void scatter2_kernel(const long long* __restrict__ ei,
                                const unsigned short* __restrict__ a2,  // lda 256
                                float* __restrict__ msg) {
    long long tid = (long long)blockIdx.x * blockDim.x + threadIdx.x;
    if (tid >= (long long)N_EDGES * 32) return;          // 32 quads of 4ch
    int e = (int)(tid >> 5), q = (int)(tid & 31);
    int src = (int)ei[e];
    int dst = (int)ei[N_EDGES + e];
    const uint2 v = *(const uint2*)&a2[(size_t)src * 256 + q * 4];
    float* p = &msg[(size_t)dst * HID_C + q * 4];
    atomicAdd(p + 0, bf2f((unsigned short)(v.x & 0xFFFF)));
    atomicAdd(p + 1, bf2f((unsigned short)(v.x >> 16)));
    atomicAdd(p + 2, bf2f((unsigned short)(v.y & 0xFFFF)));
    atomicAdd(p + 3, bf2f((unsigned short)(v.y >> 16)));
}

// ---------------- A-matrix prep ----------------
// A1[i, 0:64] = bf(x), A1[i, 64:128] = bf(msg1 * deg_inv)
__global__ void prep_a1(const float* __restrict__ x, const float* __restrict__ msg,
                        const float* __restrict__ dinv, unsigned short* __restrict__ a1) {
    long long idx = (long long)blockIdx.x * blockDim.x + threadIdx.x;
    if (idx >= (long long)N_NODES * 128) return;
    int i = (int)(idx >> 7), c = (int)(idx & 127);
    float f = (c < 64) ? x[(size_t)i * 64 + c]
                       : msg[(size_t)i * 64 + (c - 64)] * dinv[i];
    a1[idx] = f2bf(f);
}
// A2[i, 128:256] = bf(msg2 * deg_inv)   (cols 0:128 already written by gemm1)
__global__ void prep_a2(const float* __restrict__ msg, const float* __restrict__ dinv,
                        unsigned short* __restrict__ a2) {
    long long idx = (long long)blockIdx.x * blockDim.x + threadIdx.x;
    if (idx >= (long long)N_NODES * 128) return;
    int i = (int)(idx >> 7), c = (int)(idx & 127);
    a2[(size_t)i * 256 + 128 + c] = f2bf(msg[(size_t)i * 128 + c] * dinv[i]);
}

// ---- weight prep: pack [wA ; wB] (ktot x validCols, f32) into WMMA-fragment-
// ---- native bf16 layout. One dword per (kbIdx, ct, lane, vgpr) slot:
// ---- pb[((kbIdx*nct + ct)*32 + lane)*8 + v] = {B[k][n], B[k+1][n]} with
// ---- n = ct*16 + lane%16, k = kbIdx*32 + (lane/16)*16 + 2v. Cols >= validCols -> 0.
__global__ void prep_bpack(const float* __restrict__ wA, const float* __restrict__ wB,
                           unsigned int* __restrict__ pb,
                           int rowsHalf, int validCols, int ktot, int nct) {
    int idx = blockIdx.x * blockDim.x + threadIdx.x;
    int total = (ktot / 32) * nct * 32 * 8;
    if (idx >= total) return;
    int v = idx & 7;
    int lane = (idx >> 3) & 31;
    int rest = idx >> 8;
    int ct = rest % nct;
    int kbIdx = rest / nct;
    int n = ct * 16 + (lane & 15);
    int k = kbIdx * 32 + (lane >> 4) * 16 + v * 2;

    unsigned int lo = 0, hi = 0;
    if (n < validCols) {
        float f0 = (k < rowsHalf) ? wA[k * validCols + n]
                                  : wB[(k - rowsHalf) * validCols + n];
        float f1 = (k + 1 < rowsHalf) ? wA[(k + 1) * validCols + n]
                                      : wB[(k + 1 - rowsHalf) * validCols + n];
        lo = f2bf(f0);
        hi = f2bf(f1);
    }
    pb[idx] = lo | (hi << 16);
}

// ---------------- WMMA GEMM: one wave per ROW tile, NCT column tiles ----------------
// A: bf16 [rows x KTOT], row stride lda (elements). PB: pre-packed B fragments.
// D = act(A @ B + bias); bf16 or f32 output, columns masked to < storeColMax.
template <int KTOT, int NCT, bool BF16OUT, bool RELU>
__global__ __launch_bounds__(256)
void gemm_wmma(const unsigned short* __restrict__ A, int lda,
               const unsigned int* __restrict__ PB,
               const float* __restrict__ bias,
               void* __restrict__ Dst, int ldd, int storeColMax, int rowTiles) {
    int wave = threadIdx.x >> 5;
    int lane = threadIdx.x & 31;
    int tile = blockIdx.x * (blockDim.x >> 5) + wave;
    if (tile >= rowTiles) return;                   // wave-uniform; EXEC stays full
    int m0 = tile << 4;
    int half = lane >> 4;
    int mrow = m0 + (lane & 15);

    v8f acc[NCT];
#pragma unroll
    for (int ct = 0; ct < NCT; ++ct) acc[ct] = v8f{0.f,0.f,0.f,0.f,0.f,0.f,0.f,0.f};

    const unsigned short* arow = A + (size_t)mrow * lda + half * 16;
#pragma unroll
    for (int kbIdx = 0; kbIdx < KTOT / 32; ++kbIdx) {
        FragAB a;
        *(uint4*)&a.u[0] = *(const uint4*)(arow + kbIdx * 32);
        *(uint4*)&a.u[4] = *(const uint4*)(arow + kbIdx * 32 + 8);
#pragma unroll
        for (int ct = 0; ct < NCT; ++ct) {
            FragAB b;
            const unsigned int* bp = PB + ((size_t)(kbIdx * NCT + ct) * 32 + lane) * 8;
            *(uint4*)&b.u[0] = *(const uint4*)(bp);
            *(uint4*)&b.u[4] = *(const uint4*)(bp + 4);
            acc[ct] = __builtin_amdgcn_wmma_f32_16x16x32_bf16(
                false, a.v, false, b.v, (short)0, acc[ct], false, false);
        }
    }

#pragma unroll
    for (int ct = 0; ct < NCT; ++ct) {
        int n = ct * 16 + (lane & 15);
        if (n < storeColMax) {
            float bv = bias[n];
#pragma unroll
            for (int i = 0; i < 8; ++i) {
                int mm = m0 + (half << 3) + i;
                float val = acc[ct][i] + bv;
                if (RELU) val = fmaxf(val, 0.0f);
                if (BF16OUT) ((unsigned short*)Dst)[(size_t)mm * ldd + n] = f2bf(val);
                else         ((float*)Dst)[(size_t)mm * ldd + n] = val;
            }
        }
    }
}

extern "C" void kernel_launch(void* const* d_in, const int* in_sizes, int n_in,
                              void* d_out, int out_size, void* d_ws, size_t ws_size,
                              hipStream_t stream) {
    const float*     x   = (const float*)d_in[0];
    const long long* ei  = (const long long*)d_in[1];   // int64 [2, N_EDGES]
    const float* w_self1 = (const float*)d_in[2];
    const float* w_neigh1= (const float*)d_in[3];
    const float* b1      = (const float*)d_in[4];
    const float* w_self2 = (const float*)d_in[5];
    const float* w_neigh2= (const float*)d_in[6];
    const float* b2      = (const float*)d_in[7];
    const float* wc      = (const float*)d_in[8];
    const float* bc      = (const float*)d_in[9];
    float* out = (float*)d_out;

    char* ws = (char*)d_ws;
    const size_t MB = 1024 * 1024;
    float*          deg = (float*)(ws + 0);                       // 0.4 MB
    float*          msg = (float*)(ws + 1 * MB);                  // 51.2 MB (both layers)
    unsigned short* A1  = (unsigned short*)(ws + 53 * MB);        // N x 128 bf16, 25.6 MB
    unsigned short* A2  = (unsigned short*)(ws + 79 * MB);        // N x 256 bf16, 51.2 MB
    unsigned short* A3  = (unsigned short*)(ws + 131 * MB);       // N x 64 bf16, 12.8 MB
    unsigned int*   PB1 = (unsigned int*)(ws + 144 * MB);                  // 32 KB
    unsigned int*   PB2 = (unsigned int*)(ws + 144 * MB + 64 * 1024);      // 32 KB
    unsigned int*   PBC = (unsigned int*)(ws + 144 * MB + 128 * 1024);     // 4 KB

    const int T = 256;
    const int rowTiles = N_NODES / 16;                   // 6250
    const int gemmBlocks = (rowTiles + 7) / 8;           // 8 waves/block

    // weights -> packed bf16 WMMA fragments
    prep_bpack<<<(4 * 8 * 32 * 8 + T - 1) / T, T, 0, stream>>>(
        w_self1, w_neigh1, PB1, 64, HID_C, 2 * IN_C, HID_C / 16);
    prep_bpack<<<(8 * 4 * 32 * 8 + T - 1) / T, T, 0, stream>>>(
        w_self2, w_neigh2, PB2, 128, OUT_C, 2 * HID_C, OUT_C / 16);
    prep_bpack<<<(2 * 2 * 32 * 8 + T - 1) / T, T, 0, stream>>>(
        wc, wc, PBC, 64, N_CLS, OUT_C, 2);

    // degree
    hipMemsetAsync(deg, 0, (size_t)N_NODES * 4, stream);
    deg_kernel<<<(N_EDGES + T - 1) / T, T, 0, stream>>>(ei, deg);
    deg_inv_kernel<<<(N_NODES + T - 1) / T, T, 0, stream>>>(deg);

    // ---- layer 1 ----
    hipMemsetAsync(msg, 0, (size_t)N_NODES * IN_C * 4, stream);
    scatter1_kernel<<<(int)(((long long)N_EDGES * 16 + T - 1) / T), T, 0, stream>>>(ei, x, msg);
    prep_a1<<<(int)(((long long)N_NODES * 128 + T - 1) / T), T, 0, stream>>>(x, msg, deg, A1);
    // h1 = relu(A1 @ B1 + b1) written as bf16 into A2[:, 0:128]
    gemm_wmma<128, 8, true, true><<<gemmBlocks, T, 0, stream>>>(
        A1, HID_C, PB1, b1, A2, 256, HID_C, rowTiles);

    // ---- layer 2 ----
    hipMemsetAsync(msg, 0, (size_t)N_NODES * HID_C * 4, stream);
    scatter2_kernel<<<(int)(((long long)N_EDGES * 32 + T - 1) / T), T, 0, stream>>>(ei, A2, msg);
    prep_a2<<<(int)(((long long)N_NODES * 128 + T - 1) / T), T, 0, stream>>>(msg, deg, A2);
    // h2 = relu(A2 @ B2 + b2) written as bf16 into A3
    gemm_wmma<256, 4, true, true><<<gemmBlocks, T, 0, stream>>>(
        A2, 256, PB2, b2, A3, OUT_C, OUT_C, rowTiles);

    // ---- classifier: out = A3 @ BC + bc (cols masked to 20) ----
    gemm_wmma<64, 2, false, false><<<gemmBlocks, T, 0, stream>>>(
        A3, OUT_C, PBC, bc, out, N_CLS, N_CLS, rowTiles);
}
